// Interaction_MLP4d_layer_36086315221299
// MI455X (gfx1250) — matrix operand order repaired and checked by reading, not compiled, not run on hardware
//
#include <hip/hip_runtime.h>
#include <math.h>

// Problem constants (fixed by the reference)
#define NN    64          // nodes
#define BB    8           // batch
#define TT    64          // feature len
#define HH    32          // hidden
#define EE    4032        // edges = N*(N-1), grouped by fst (permutations order)
#define DEG   63          // edges per source node
#define ESTRIPE 256
#define NSTRIPE 64

typedef __attribute__((ext_vector_type(16))) _Float16 v16h;
typedef __attribute__((ext_vector_type(8)))  _Float16 v8h;
typedef __attribute__((ext_vector_type(8)))  float    v8f;

// Fast silu: x * rcp(1 + exp(-x)).  Output feeds BatchNorm (and operands were
// already f16-rounded), so v_rcp_f32 accuracy (~1 ulp) is more than enough and
// avoids the precise-division v_div_scale/v_div_fmas expansion.
__device__ __forceinline__ float siluf(float x) {
  return x * __builtin_amdgcn_rcpf(1.0f + __expf(-x));
}

// CDNA5 wave32 A-fragment (16-bit A, 16x32): halfs 0..7 = A[m][kbase + 8*half .. +7],
// halfs 8..15 = A[m][kbase + 16 + 8*half .. +7]  (per ISA 7.12.2 table)
__device__ __forceinline__ v16h load_a_frag(const _Float16* rowk, int half) {
  v8h lo = *(const v8h*)(rowk + half * 8);
  v8h hi = *(const v8h*)(rowk + 16 + half * 8);
  v16h a;
#pragma unroll
  for (int i = 0; i < 8; ++i) { a[i] = lo[i]; a[i + 8] = hi[i]; }
  return a;
}

// B-fragment from weights stored transposed in LDS (WT[n][k]):
// lane supplies n = lane%16; halfs 0..7 = B[kbase + 8*half + idx][n], halfs 8..15 = B[kbase + 16 + 8*half + idx-8][n]
__device__ __forceinline__ v16h load_b_frag(const _Float16* rowk, int half) {
  v8h lo = *(const v8h*)(rowk + half * 8);
  v8h hi = *(const v8h*)(rowk + 16 + half * 8);
  v16h b;
#pragma unroll
  for (int i = 0; i < 8; ++i) { b[i] = lo[i]; b[i + 8] = hi[i]; }
  return b;
}

// Shared-memory byte offsets (58240 bytes total; Amat region is reused for S)
#define SM_AMAT 0        // 64x64 f16 (8192 B)   -- also S: 64x32 f32 (8192 B)
#define SM_WDW  8192     // 32x64 f16 (4096 B)   W1^T  [h][k]
#define SM_Y    12288    // 32x64 f16 (4096 B)   silu(L1)^T [h][t]
#define SM_WF1  16384    // 128x64 f16 (16384 B) W2^T  [u][k=t]
#define SM_Z    32768    // 32x128 f16 (8192 B)  silu(L2) [h][u]
#define SM_WF2  40960    // 64x128 f16 (16384 B) W3^T  [t][k=u]
#define SM_DB   57344    // 32 f32
#define SM_F1B  57472    // 128 f32
#define SM_F2B  57984    // 64 f32
#define SM_SIZE 58240

// Fused 3-layer MLP on one [64t x 64k] tile already staged in Amat.
// Produces pre-BN S[t][h] (f32) in the Amat region. 8 waves, 256 threads.
__device__ __forceinline__ void mlp_core(char* smem, int tid) {
  _Float16* Amat = (_Float16*)(smem + SM_AMAT);
  _Float16* Wdw  = (_Float16*)(smem + SM_WDW);
  _Float16* Y    = (_Float16*)(smem + SM_Y);
  _Float16* Wf1  = (_Float16*)(smem + SM_WF1);
  _Float16* Z    = (_Float16*)(smem + SM_Z);
  _Float16* Wf2  = (_Float16*)(smem + SM_WF2);
  float* dbL  = (float*)(smem + SM_DB);
  float* f1bL = (float*)(smem + SM_F1B);
  float* f2bL = (float*)(smem + SM_F2B);
  float* S    = (float*)(smem + SM_AMAT);

  const int lane = tid & 31, wave = tid >> 5;
  const int half = lane >> 4, lm = lane & 15;

  // ---- Stage 1: D1[t=64][h=32] = Amat[64x64] * Wdw ; 4x2 tiles, 1 per wave
  {
    const int mtile = wave >> 1, ntile = wave & 1;
    const int mrow = mtile * 16 + lm;   // t row (A)
    const int ncol = ntile * 16 + lm;   // h col (B)
    v8f c = {};
#pragma unroll
    for (int kt = 0; kt < 2; ++kt) {
      v16h a  = load_a_frag(Amat + mrow * 64 + kt * 32, half);
      v16h bb = load_b_frag(Wdw  + ncol * 64 + kt * 32, half);
      c = __builtin_amdgcn_wmma_f32_16x16x32_f16(false, a, false, bb,
                                                 (short)0, c, false, false);
    }
    const float bias = dbL[ncol];
#pragma unroll
    for (int v = 0; v < 8; ++v) {
      const int trow = mtile * 16 + half * 8 + v;     // D m-index
      Y[ncol * 64 + trow] = (_Float16)siluf(c[v] + bias);  // transpose -> Y[h][t]
    }
  }
  __syncthreads();

  // ---- Stage 2: D2[h=32][u=128] = Y[32x64] * Wf1 ; 2x8 tiles, 2 per wave
  {
    const int mtile = wave >> 2;
    const int nt0 = (wave & 3) * 2;
#pragma unroll
    for (int nn = 0; nn < 2; ++nn) {
      const int ntile = nt0 + nn;
      const int mrow = mtile * 16 + lm;  // h
      const int ncol = ntile * 16 + lm;  // u
      v8f c = {};
#pragma unroll
      for (int kt = 0; kt < 2; ++kt) {
        v16h a  = load_a_frag(Y   + mrow * 64 + kt * 32, half);
        v16h bb = load_b_frag(Wf1 + ncol * 64 + kt * 32, half);
        c = __builtin_amdgcn_wmma_f32_16x16x32_f16(false, a, false, bb,
                                                   (short)0, c, false, false);
      }
      const float bias = f1bL[ncol];
#pragma unroll
      for (int v = 0; v < 8; ++v) {
        const int hrow = mtile * 16 + half * 8 + v;
        Z[hrow * 128 + ncol] = (_Float16)siluf(c[v] + bias);  // Z[h][u]
      }
    }
  }
  __syncthreads();

  // ---- Stage 3: D3[h=32][t=64] = Z[32x128] * Wf2 ; 2x4 tiles, 1 per wave
  {
    const int mtile = wave >> 2, ntile = wave & 3;
    const int mrow = mtile * 16 + lm;  // h
    const int ncol = ntile * 16 + lm;  // t
    v8f c = {};
#pragma unroll
    for (int kt = 0; kt < 4; ++kt) {
      v16h a  = load_a_frag(Z   + mrow * 128 + kt * 32, half);
      v16h bb = load_b_frag(Wf2 + ncol * 128 + kt * 32, half);
      c = __builtin_amdgcn_wmma_f32_16x16x32_f16(false, a, false, bb,
                                                 (short)0, c, false, false);
    }
    const float bias = f2bL[ncol];
#pragma unroll
    for (int v = 0; v < 8; ++v) {
      const int hrow = mtile * 16 + half * 8 + v;
      S[ncol * 32 + hrow] = siluf(c[v] + bias);  // S[t][h] (pre-BN, f32)
    }
  }
  __syncthreads();
}

__device__ __forceinline__ void stage_weights(char* smem,
    const float* dw, const float* db, const float* f1w, const float* f1b,
    const float* f2w, const float* f2b, int tid) {
  _Float16* Wdw = (_Float16*)(smem + SM_WDW);
  _Float16* Wf1 = (_Float16*)(smem + SM_WF1);
  _Float16* Wf2 = (_Float16*)(smem + SM_WF2);
  float* dbL  = (float*)(smem + SM_DB);
  float* f1bL = (float*)(smem + SM_F1B);
  float* f2bL = (float*)(smem + SM_F2B);
  for (int i = tid; i < 32 * 64; i += 256)  { int h = i >> 6, k = i & 63;  Wdw[i] = (_Float16)dw[k * 32 + h]; }
  for (int i = tid; i < 128 * 64; i += 256) { int u = i >> 6, k = i & 63;  Wf1[i] = (_Float16)f1w[k * 128 + u]; }
  for (int i = tid; i < 64 * 128; i += 256) { int t = i >> 7, k = i & 127; Wf2[i] = (_Float16)f2w[k * 64 + t]; }
  if (tid < 32)  dbL[tid]  = db[tid];
  if (tid < 128) f1bL[tid] = f1b[tid];
  if (tid < 64)  f2bL[tid] = f2b[tid];
}

// ---------------- Edge MLP: one block per (e,b). Writes pre-BN x3 to out2
// at final layout [e][b][t][h]; accumulates striped BN stats.
__global__ __launch_bounds__(256) void k_edge_mlp(
    const float* __restrict__ nf, const int* __restrict__ eidx,
    const float* __restrict__ dw, const float* __restrict__ db,
    const float* __restrict__ f1w, const float* __restrict__ f1b,
    const float* __restrict__ f2w, const float* __restrict__ f2b,
    float* __restrict__ out2, float* __restrict__ stats) {
  __shared__ __align__(64) char smem[SM_SIZE];
  const int tid = threadIdx.x;
  const int e = blockIdx.x >> 3, b = blockIdx.x & 7;
  const int fst = eidx[e], snd = eidx[EE + e];

  stage_weights(smem, dw, db, f1w, f1b, f2w, f2b, tid);

  // Amat[t][k]: k<32 from x[snd], k>=32 from x[fst]; x[n,b,t,i]=nf[((b*32+i)*64+n)*64+t]
  _Float16* Amat = (_Float16*)(smem + SM_AMAT);
  for (int i = tid; i < 64 * 64; i += 256) {
    const int k = i >> 6, t = i & 63;
    const int nsrc = (k < 32) ? snd : fst;
    Amat[t * 64 + k] = (_Float16)nf[(((b * 32 + (k & 31)) * 64 + nsrc) << 6) + t];
  }
  __syncthreads();

  mlp_core(smem, tid);

  const float* S = (const float*)(smem + SM_AMAT);
  const int base = blockIdx.x << 11;  // (e*8+b)*2048, layout [t][h] matches S
  for (int i = tid; i < 2048; i += 256) out2[base + i] = S[i];

  if (tid < 64) {  // per-t partial sums over 32 h values
    float s = 0.f, ss = 0.f;
#pragma unroll
    for (int h = 0; h < 32; ++h) { float v = S[tid * 32 + h]; s += v; ss += v * v; }
    float* st = stats + (blockIdx.x & (ESTRIPE - 1)) * 128;
    atomicAdd(st + tid, s);
    atomicAdd(st + 64 + tid, ss);
  }
}

// ---------------- Node MLP: one block per (n,b). Writes pre-BN to out1 [b][h][n][t].
__global__ __launch_bounds__(256) void k_node_mlp(
    const float* __restrict__ nf, const float* __restrict__ xadj,
    const float* __restrict__ dw, const float* __restrict__ db,
    const float* __restrict__ f1w, const float* __restrict__ f1b,
    const float* __restrict__ f2w, const float* __restrict__ f2b,
    float* __restrict__ out1, float* __restrict__ stats) {
  __shared__ __align__(64) char smem[SM_SIZE];
  const int tid = threadIdx.x;
  const int n = blockIdx.x >> 3, b = blockIdx.x & 7;

  stage_weights(smem, dw, db, f1w, f1b, f2w, f2b, tid);

  _Float16* Amat = (_Float16*)(smem + SM_AMAT);
  for (int i = tid; i < 64 * 64; i += 256) {
    const int k = i >> 6, t = i & 63;
    const float v = (k < 32)
        ? nf[(((b * 32 + k) * 64 + n) << 6) + t]
        : xadj[((n * 8 + b) << 11) + t * 32 + (k - 32)];
    Amat[t * 64 + k] = (_Float16)v;
  }
  __syncthreads();

  mlp_core(smem, tid);

  const float* S = (const float*)(smem + SM_AMAT);
  for (int i = tid; i < 2048; i += 256) {
    const int h = i >> 6, t = i & 63;
    out1[(((b * 32 + h) * 64 + n) << 6) + t] = S[t * 32 + h];
  }
  if (tid < 64) {
    float s = 0.f, ss = 0.f;
#pragma unroll
    for (int h = 0; h < 32; ++h) { float v = S[tid * 32 + h]; s += v; ss += v * v; }
    float* st = stats + (blockIdx.x & (NSTRIPE - 1)) * 128;
    atomicAdd(st + tid, s);
    atomicAdd(st + 64 + tid, ss);
  }
}

// ---------------- BN stat finalize: AB[t]=inv_std*g, AB[64+t]=b2 - mean*AB[t]
__global__ void k_finalize(const float* __restrict__ stats, int nstripe, float count,
                           const float* __restrict__ g, const float* __restrict__ b2,
                           float* __restrict__ AB) {
  const int t = threadIdx.x;
  if (t >= 64) return;
  float s = 0.f, ss = 0.f;
  for (int j = 0; j < nstripe; ++j) { s += stats[j * 128 + t]; ss += stats[j * 128 + 64 + t]; }
  const float m = s / count;
  const float var = ss / count - m * m;       // biased var (jnp.var)
  const float A = rsqrtf(var + 1e-5f) * g[t];
  AB[t] = A;
  AB[64 + t] = b2[t] - m * A;
}

// ---------------- Fused in-place BN + edges_mask + segment-mean on out2.
// One block per (n,b): the 63 edges with fst==n are a contiguous, exclusively
// owned slice of out2 ([e][b][t][h]), so normalize in place and accumulate the
// scatter-mean in a single pass (saves a full 264MB read at 23.3 TB/s).
__global__ __launch_bounds__(256) void k_bn_agg(
    float* __restrict__ out2, const float* __restrict__ AB,
    const float* __restrict__ emask, float* __restrict__ xadj) {
  __shared__ float sA[64], sB[64], em[DEG];
  const int tid = threadIdx.x;
  const int n = blockIdx.x >> 3, b = blockIdx.x & 7;
  if (tid < 64) { sA[tid] = AB[tid]; sB[tid] = AB[64 + tid]; }
  if (tid < DEG) em[tid] = emask[b * EE + n * DEG + tid];
  __syncthreads();
  float* base = out2 + (((size_t)(n * DEG) * 8 + b) << 11);
  for (int i = tid; i < 2048; i += 256) {
    const int t = (i >> 5) & 63;
    const float a = sA[t], c = sB[t];
    float acc = 0.f;
    float* p = base + i;
#pragma unroll 7
    for (int j = 0; j < DEG; ++j) {
      float v = (p[(size_t)j << 14] * a + c) * em[j];  // j*(8*2048)
      p[(size_t)j << 14] = v;
      acc += v;
    }
    xadj[(((n << 3) + b) << 11) + i] = acc * (1.0f / (float)DEG);
  }
}

// ---------------- In-place BN + nodes_mask on out1 ([b][h][n][t])
__global__ void k_bn_node(float* __restrict__ out1, const float* __restrict__ AB,
                          const float* __restrict__ nmask) {
  __shared__ float sA[64], sB[64];
  const int tid = threadIdx.x;
  if (tid < 64) { sA[tid] = AB[tid]; sB[tid] = AB[64 + tid]; }
  __syncthreads();
  const int base = blockIdx.x * 2048;
  const int b = base >> 17;  // constant per block
#pragma unroll
  for (int j = 0; j < 8; ++j) {
    const int idx = base + j * 256 + tid;
    const int t = idx & 63, n = (idx >> 6) & 63;
    out1[idx] = (out1[idx] * sA[t] + sB[t]) * nmask[b * 64 + n];
  }
}

__global__ void k_zero(float* __restrict__ p, int n) {
  for (int i = blockIdx.x * 256 + threadIdx.x; i < n; i += gridDim.x * 256) p[i] = 0.f;
}

extern "C" void kernel_launch(void* const* d_in, const int* in_sizes, int n_in,
                              void* d_out, int out_size, void* d_ws, size_t ws_size,
                              hipStream_t stream) {
  const float* nf    = (const float*)d_in[0];   // node_feature [B,DIN,N,T]
  const float* nmask = (const float*)d_in[2];   // [B,N]
  const float* emask = (const float*)d_in[3];   // [B,E]
  const float* e_dw  = (const float*)d_in[4];
  const float* e_db  = (const float*)d_in[5];
  const float* e_f1w = (const float*)d_in[6];
  const float* e_f1b = (const float*)d_in[7];
  const float* e_f2w = (const float*)d_in[8];
  const float* e_f2b = (const float*)d_in[9];
  const float* e_g   = (const float*)d_in[10];
  const float* e_b2  = (const float*)d_in[11];
  const float* n_dw  = (const float*)d_in[12];
  const float* n_db  = (const float*)d_in[13];
  const float* n_f1w = (const float*)d_in[14];
  const float* n_f1b = (const float*)d_in[15];
  const float* n_f2w = (const float*)d_in[16];
  const float* n_f2b = (const float*)d_in[17];
  const float* n_g   = (const float*)d_in[18];
  const float* n_b2  = (const float*)d_in[19];
  const int*   eidx  = (const int*)d_in[20];    // [2,E]

  float* out1 = (float*)d_out;                       // [B,H,N,T] = 1,048,576 f32
  float* out2 = out1 + (size_t)BB * HH * NN * TT;    // [E,B,T,H] = 66,060,288 f32

  float* ws     = (float*)d_ws;
  float* eStats = ws;                 // 256*128 = 32768
  float* nStats = ws + 32768;         // 64*128 = 8192
  float* eAB    = ws + 40960;         // 128
  float* nAB    = ws + 41088;         // 128
  float* xadj   = ws + 41216;         // N*B*T*H = 1,048,576

  k_zero<<<160, 256, 0, stream>>>(eStats, 40960);

  k_edge_mlp<<<EE * BB, 256, 0, stream>>>(nf, eidx, e_dw, e_db, e_f1w, e_f1b,
                                          e_f2w, e_f2b, out2, eStats);
  k_finalize<<<1, 64, 0, stream>>>(eStats, ESTRIPE,
                                   (float)((size_t)BB * HH * EE), e_g, e_b2, eAB);
  k_bn_agg<<<NN * BB, 256, 0, stream>>>(out2, eAB, emask, xadj);

  k_node_mlp<<<NN * BB, 256, 0, stream>>>(nf, xadj, n_dw, n_db, n_f1w, n_f1b,
                                          n_f2w, n_f2b, out1, nStats);
  k_finalize<<<1, 64, 0, stream>>>(nStats, NSTRIPE,
                                   (float)((size_t)BB * HH * NN), n_g, n_b2, nAB);
  k_bn_node<<<NN * BB, 256, 0, stream>>>(out1, nAB, nmask);
}
